// MultiHeadAttention_62586263437953
// MI455X (gfx1250) — compile-verified, matrix operand order checked by reading, not run on hardware
//
#include <hip/hip_runtime.h>
#include <hip/hip_bf16.h>

// ---------------------------------------------------------------------------
// MHA forward for MI455X (gfx1250, wave32, WMMA).
// B=2, S=2048, D=1024, H=16, dk=64.
// d_out = [ out (B*S*D f32) | attn (B*H*S*S f32) ]
// d_ws  = [ q16 | k16 | v16T | ctx16 ]  (f16, 8MB each = 32MB total)
// ---------------------------------------------------------------------------

typedef __attribute__((ext_vector_type(16))) _Float16 v16h;
typedef __attribute__((ext_vector_type(8)))  _Float16 v8h;
typedef __attribute__((ext_vector_type(8)))  float    v8f;

#define B_  2
#define S_  2048
#define D_  1024
#define H_  16
#define DK_ 64

// Tile config: workgroup = 256 threads = 8 waves (wave32).
// Output tile 128(M) x 64(N); waves arranged 4(M) x 2(N), each wave 32x32
// (= 2x2 WMMA 16x16 accumulators). K-step = 32 (one v_wmma_f32_16x16x32_f16).
#define BM 128
#define BN 64
#define BK 32
#define LDS_STRIDE 40   // 32 + 8 halfs pad (16B-aligned) to spread LDS banks

// --- LDS fragment load --------------------------------------------------
// ISA 7.12.2, 16-bit A 16x32: lane = m%16, half = lane/16.
// half h covers K in {h*8 .. h*8+7} (VGPR0-3) and {16+h*8 .. 16+h*8+7} (VGPR4-7),
// packed pairwise little-endian -> element i of v16h:
//   i<8 : K = h*8 + i          (contiguous 16B)
//   i>=8: K = 16 + h*8 + (i-8) (contiguous 16B)
// => two ds_load_b128 per fragment. B assumed layout-dual to A (lane = n%16).
__device__ __forceinline__ v16h frag_ld(const _Float16* lds, int row, int half) {
    const _Float16* p = lds + row * LDS_STRIDE;
    v8h lo = *(const v8h*)(p + half * 8);
    v8h hi = *(const v8h*)(p + 16 + half * 8);
    v16h f;
#pragma unroll
    for (int i = 0; i < 8; ++i) { f[i] = lo[i]; f[i + 8] = hi[i]; }
    return f;
}

// One BK step for a wave: 2 A frags, 2 B frags, 4 WMMAs.
__device__ __forceinline__ void wave_mma(const _Float16* Alds, const _Float16* Blds,
                                         int wm, int wn, int lane, v8f (&acc)[2][2]) {
    const int lrow = lane & 15, half = lane >> 4;
    v16h fa[2], fb[2];
#pragma unroll
    for (int mi = 0; mi < 2; ++mi) fa[mi] = frag_ld(Alds, wm * 32 + mi * 16 + lrow, half);
#pragma unroll
    for (int ni = 0; ni < 2; ++ni) fb[ni] = frag_ld(Blds, wn * 32 + ni * 16 + lrow, half);
#pragma unroll
    for (int mi = 0; mi < 2; ++mi)
#pragma unroll
        for (int ni = 0; ni < 2; ++ni)
            acc[mi][ni] = __builtin_amdgcn_wmma_f32_16x16x32_f16(
                false, fa[mi], false, fb[ni], (short)0, acc[mi][ni], false, false);
}

// --- global -> LDS staging ----------------------------------------------
// f32 source, convert to f16. src pre-offset to (row0, k0). rows*32 elems.
__device__ __forceinline__ void stage_f32(const float* __restrict__ src, int ld,
                                          int rows, _Float16* dst, int tid) {
    const int total = rows * 32;
    for (int idx = tid * 4; idx < total; idx += 256 * 4) {
        const int r = idx >> 5, c = idx & 31;
        const float4 v = *(const float4*)(src + (size_t)r * ld + c);
        _Float16* d = dst + r * LDS_STRIDE + c;
        d[0] = (_Float16)v.x; d[1] = (_Float16)v.y;
        d[2] = (_Float16)v.z; d[3] = (_Float16)v.w;
    }
}

// f16 source, straight copy (16B vector moves).
__device__ __forceinline__ void stage_f16(const _Float16* __restrict__ src, int ld,
                                          int rows, _Float16* dst, int tid) {
    const int total = rows * 32;
    for (int idx = tid * 8; idx < total; idx += 256 * 8) {
        const int r = idx >> 5, c = idx & 31;
        *(v8h*)(dst + r * LDS_STRIDE + c) = *(const v8h*)(src + (size_t)r * ld + c);
    }
}

// ===========================================================================
// Kernel 1: fused QKV projections.  C[m,n] = X[m,:] . W[n,:] + b[n]
// (torch Linear x@W.T: both X and W stream row-major along k -> no transposes)
// q,k stored [B,H,S,dk] f16; v stored transposed [B,H,dk,S] f16 so the
// ctx GEMM's B-fragments are contiguous rows.
// ===========================================================================
__global__ void __launch_bounds__(256) qkv_proj_kernel(
    const float* __restrict__ Q, const float* __restrict__ K, const float* __restrict__ V,
    const float* __restrict__ Wq, const float* __restrict__ bq,
    const float* __restrict__ Wk, const float* __restrict__ bk,
    const float* __restrict__ Wv, const float* __restrict__ bv,
    _Float16* __restrict__ q16, _Float16* __restrict__ k16, _Float16* __restrict__ v16t)
{
    __shared__ _Float16 Alds[BM * LDS_STRIDE];
    __shared__ _Float16 Blds[BN * LDS_STRIDE];
    const int z = blockIdx.z;
    const float* X    = (z == 0) ? Q  : (z == 1) ? K  : V;
    const float* W    = (z == 0) ? Wq : (z == 1) ? Wk : Wv;
    const float* bias = (z == 0) ? bq : (z == 1) ? bk : bv;
    const int m0 = blockIdx.x * BM, n0 = blockIdx.y * BN;
    const int tid = threadIdx.x, waveId = tid >> 5, lane = tid & 31;
    const int wm = waveId & 3, wn = waveId >> 2;

    v8f acc[2][2] = {};
    for (int k = 0; k < D_; k += BK) {
        stage_f32(X + (size_t)m0 * D_ + k, D_, BM, Alds, tid);
        stage_f32(W + (size_t)n0 * D_ + k, D_, BN, Blds, tid);
        if (k + BK < D_)  // gfx1250 speculative prefetch of next A tile
            __builtin_prefetch(X + (size_t)(m0 + (tid & 127)) * D_ + k + BK, 0, 0);
        __syncthreads();
        wave_mma(Alds, Blds, wm, wn, lane, acc);
        __syncthreads();
    }

    const int lrow = lane & 15, lhalf = lane >> 4;
#pragma unroll
    for (int ni = 0; ni < 2; ++ni) {
        const int n = n0 + wn * 32 + ni * 16 + lrow;
        const float bn = bias[n];
        const int h = n >> 6, d = n & 63;
#pragma unroll
        for (int mi = 0; mi < 2; ++mi) {
#pragma unroll
            for (int r = 0; r < 8; ++r) {
                const int m = m0 + wm * 32 + mi * 16 + lhalf * 8 + r;
                const float val = acc[mi][ni][r] + bn;
                const int b = m >> 11, s = m & (S_ - 1);
                if (z == 0)
                    q16[(((size_t)b * H_ + h) * S_ + s) * DK_ + d] = (_Float16)val;
                else if (z == 1)
                    k16[(((size_t)b * H_ + h) * S_ + s) * DK_ + d] = (_Float16)val;
                else
                    v16t[(((size_t)b * H_ + h) * DK_ + d) * S_ + s] = (_Float16)val;
            }
        }
    }
}

// ===========================================================================
// Kernel 2: scores[i,j] = (q[i,:].k[j,:]) / 8, masked -> attn region (f32).
// ===========================================================================
__global__ void __launch_bounds__(256) scores_kernel(
    const _Float16* __restrict__ q16, const _Float16* __restrict__ k16,
    const int* __restrict__ mask, float* __restrict__ attn)
{
    __shared__ _Float16 Alds[BM * LDS_STRIDE];
    __shared__ _Float16 Blds[BN * LDS_STRIDE];
    const int bh = blockIdx.z, b = bh >> 4;
    const int m0 = blockIdx.x * BM, n0 = blockIdx.y * BN;
    const size_t head = (size_t)bh * S_ * DK_;
    const int tid = threadIdx.x, waveId = tid >> 5, lane = tid & 31;
    const int wm = waveId & 3, wn = waveId >> 2;

    v8f acc[2][2] = {};
#pragma unroll
    for (int k = 0; k < DK_; k += BK) {
        stage_f16(q16 + head + (size_t)m0 * DK_ + k, DK_, BM, Alds, tid);
        stage_f16(k16 + head + (size_t)n0 * DK_ + k, DK_, BN, Blds, tid);
        __syncthreads();
        wave_mma(Alds, Blds, wm, wn, lane, acc);
        __syncthreads();
    }

    const int lrow = lane & 15, lhalf = lane >> 4;
#pragma unroll
    for (int ni = 0; ni < 2; ++ni) {
        const int n = n0 + wn * 32 + ni * 16 + lrow;       // key index j
#pragma unroll
        for (int mi = 0; mi < 2; ++mi) {
#pragma unroll
            for (int r = 0; r < 8; ++r) {
                const int m = m0 + wm * 32 + mi * 16 + lhalf * 8 + r;  // query i
                const float sv = acc[mi][ni][r] * 0.125f;  // 1/sqrt(64)
                const int mk = mask[((size_t)b * S_ + m) * S_ + n];
                attn[((size_t)bh * S_ + m) * S_ + n] = (mk == 0) ? -1e9f : sv;
            }
        }
    }
}

// ===========================================================================
// Kernel 3: row softmax. One wave32 per 2048-wide row; entire row lives in
// 64 VGPRs -> exactly one read + one write pass over the 537MB attn tensor.
// ===========================================================================
__global__ void __launch_bounds__(256) softmax_kernel(float* __restrict__ attn)
{
    const int wave = threadIdx.x >> 5, lane = threadIdx.x & 31;
    const size_t row = (size_t)blockIdx.x * 8 + wave;
    float* p = attn + row * S_;
    float v[64];
    float mx = -3.4e38f;
#pragma unroll
    for (int i = 0; i < 64; ++i) { v[i] = p[lane + i * 32]; mx = fmaxf(mx, v[i]); }
#pragma unroll
    for (int off = 16; off > 0; off >>= 1) mx = fmaxf(mx, __shfl_xor(mx, off, 32));
    float sum = 0.f;
#pragma unroll
    for (int i = 0; i < 64; ++i) { v[i] = __expf(v[i] - mx); sum += v[i]; }
#pragma unroll
    for (int off = 16; off > 0; off >>= 1) sum += __shfl_xor(sum, off, 32);
    const float inv = 1.0f / sum;   // fully-masked row: all exp(0) -> sum=2048, safe
#pragma unroll
    for (int i = 0; i < 64; ++i) p[lane + i * 32] = v[i] * inv;
}

// ===========================================================================
// Kernel 4: ctx = attn @ v.  N = dk = 64 -> one block column. B-fragments come
// from v^T [B,H,dk,S]: row d contiguous along k. attn f32 -> f16 on LDS stage.
// ctx stored f16, re-interleaved to [B,S,D] (head-major columns).
// ===========================================================================
__global__ void __launch_bounds__(256) ctx_kernel(
    const float* __restrict__ attn, const _Float16* __restrict__ v16t,
    _Float16* __restrict__ ctx16)
{
    __shared__ _Float16 Alds[BM * LDS_STRIDE];
    __shared__ _Float16 Blds[BN * LDS_STRIDE];
    const int bh = blockIdx.z, b = bh >> 4, h = bh & 15;
    const int m0 = blockIdx.x * BM;
    const float*    Abase = attn + (size_t)bh * S_ * S_ + (size_t)m0 * S_;
    const _Float16* Bbase = v16t + (size_t)bh * DK_ * S_;
    const int tid = threadIdx.x, waveId = tid >> 5, lane = tid & 31;
    const int wm = waveId & 3, wn = waveId >> 2;

    v8f acc[2][2] = {};
    for (int k = 0; k < S_; k += BK) {
        stage_f32(Abase + k, S_, BM, Alds, tid);
        stage_f16(Bbase + k, S_, BN, Blds, tid);
        if (k + BK < S_)
            __builtin_prefetch(Abase + (size_t)(tid & 127) * S_ + k + BK, 0, 0);
        __syncthreads();
        wave_mma(Alds, Blds, wm, wn, lane, acc);
        __syncthreads();
    }

    const int lrow = lane & 15, lhalf = lane >> 4;
#pragma unroll
    for (int ni = 0; ni < 2; ++ni) {
        const int n = wn * 32 + ni * 16 + lrow;            // dk index
#pragma unroll
        for (int mi = 0; mi < 2; ++mi) {
#pragma unroll
            for (int r = 0; r < 8; ++r) {
                const int s = m0 + wm * 32 + mi * 16 + lhalf * 8 + r;
                ctx16[((size_t)b * S_ + s) * D_ + h * DK_ + n] = (_Float16)acc[mi][ni][r];
            }
        }
    }
}

// ===========================================================================
// Kernel 5: out = ctx @ Wo^T + bo (f32 out).
// ===========================================================================
__global__ void __launch_bounds__(256) out_proj_kernel(
    const _Float16* __restrict__ ctx16, const float* __restrict__ Wo,
    const float* __restrict__ bo, float* __restrict__ out)
{
    __shared__ _Float16 Alds[BM * LDS_STRIDE];
    __shared__ _Float16 Blds[BN * LDS_STRIDE];
    const int m0 = blockIdx.x * BM, n0 = blockIdx.y * BN;
    const int tid = threadIdx.x, waveId = tid >> 5, lane = tid & 31;
    const int wm = waveId & 3, wn = waveId >> 2;

    v8f acc[2][2] = {};
    for (int k = 0; k < D_; k += BK) {
        stage_f16(ctx16 + (size_t)m0 * D_ + k, D_, BM, Alds, tid);
        stage_f32(Wo + (size_t)n0 * D_ + k, D_, BN, Blds, tid);
        if (k + BK < D_)
            __builtin_prefetch(ctx16 + (size_t)(m0 + (tid & 127)) * D_ + k + BK, 0, 0);
        __syncthreads();
        wave_mma(Alds, Blds, wm, wn, lane, acc);
        __syncthreads();
    }

    const int lrow = lane & 15, lhalf = lane >> 4;
#pragma unroll
    for (int ni = 0; ni < 2; ++ni) {
        const int n = n0 + wn * 32 + ni * 16 + lrow;
        const float bn = bo[n];
#pragma unroll
        for (int mi = 0; mi < 2; ++mi) {
#pragma unroll
            for (int r = 0; r < 8; ++r) {
                const int m = m0 + wm * 32 + mi * 16 + lhalf * 8 + r;
                out[(size_t)m * D_ + n] = acc[mi][ni][r] + bn;
            }
        }
    }
}

// ===========================================================================
extern "C" void kernel_launch(void* const* d_in, const int* in_sizes, int n_in,
                              void* d_out, int out_size, void* d_ws, size_t ws_size,
                              hipStream_t stream) {
    const float* Q    = (const float*)d_in[0];
    const float* K    = (const float*)d_in[1];
    const float* V    = (const float*)d_in[2];
    const int*   mask = (const int*)  d_in[3];
    const float* Wq = (const float*)d_in[4],  *bq = (const float*)d_in[5];
    const float* Wk = (const float*)d_in[6],  *bk = (const float*)d_in[7];
    const float* Wv = (const float*)d_in[8],  *bv = (const float*)d_in[9];
    const float* Wo = (const float*)d_in[10], *bo = (const float*)d_in[11];

    float* out  = (float*)d_out;
    float* attn = out + (size_t)B_ * S_ * D_;          // tuple: (out, attn)

    const size_t HEADSZ = (size_t)B_ * H_ * S_ * DK_;  // 4,194,304 halfs
    _Float16* ws    = (_Float16*)d_ws;
    _Float16* q16   = ws;
    _Float16* k16   = ws + HEADSZ;
    _Float16* v16t  = ws + 2 * HEADSZ;
    _Float16* ctx16 = ws + 3 * HEADSZ;                 // 32MB total

    const dim3 blk(256);
    // 1. QKV projections: M=4096, N=1024 -> (32,16,3)
    qkv_proj_kernel<<<dim3(32, 16, 3), blk, 0, stream>>>(
        Q, K, V, Wq, bq, Wk, bk, Wv, bv, q16, k16, v16t);
    // 2. scores+mask: per head M=2048, N=2048 -> (16,32,32)
    scores_kernel<<<dim3(16, 32, B_ * H_), blk, 0, stream>>>(q16, k16, mask, attn);
    // 3. softmax: 65536 rows, 8 waves/block
    softmax_kernel<<<dim3((B_ * H_ * S_) / 8), blk, 0, stream>>>(attn);
    // 4. ctx = attn @ v: per head M=2048, N=64 -> (16,1,32)
    ctx_kernel<<<dim3(16, 1, B_ * H_), blk, 0, stream>>>(attn, v16t, ctx16);
    // 5. output projection: M=4096, N=1024 -> (32,16)
    out_proj_kernel<<<dim3(32, 16, 1), blk, 0, stream>>>(ctx16, Wo, bo, out);
}
